// ProgressiveExpandableFFN_12146167513167
// MI455X (gfx1250) — matrix operand order repaired and checked by reading, not compiled, not run on hardware
//
#include <hip/hip_runtime.h>
#include <cstdint>

typedef _Float16 f16;
typedef __attribute__((ext_vector_type(16))) _Float16 v16h;
typedef __attribute__((ext_vector_type(8)))  _Float16 v8h;
typedef __attribute__((ext_vector_type(4)))  _Float16 v4h;
typedef __attribute__((ext_vector_type(8)))  float    v8f;

#define DM   1024      // d_model
#define DF   4096      // d_ff
#define NTOK 8192      // 4 * 2048 tokens
#define BM   128
#define BN   128
#define BK   64
#define LDA  72        // padded f16 elems per LDS row (144B: 16B aligned)

__device__ __forceinline__ float fake_q(float v, float s, float inv_s, float lo, float hi) {
    float q = rintf(v * s);                 // round-half-even, matches jnp.round
    q = fminf(fmaxf(q, lo), hi);
    return q * inv_s;
}

// LDS byte offset of a shared-memory pointer (ptrtoint of addrspace(3))
__device__ __forceinline__ unsigned lds_off(const void* p) {
    return (unsigned)(unsigned long long)(__attribute__((address_space(3))) const char*)p;
}

// CDNA5 async global->LDS copy, 16B per lane, tracked by ASYNCcnt
__device__ __forceinline__ void async_b128(unsigned lds_byte_off, const void* gptr) {
    asm volatile("global_load_async_to_lds_b128 %0, %1, off"
                 :: "v"(lds_byte_off), "v"((unsigned long long)gptr)
                 : "memory");
}
__device__ __forceinline__ void wait_async() {
    asm volatile("s_wait_asynccnt 0" ::: "memory");
}

// ---------------- scale / quant prep kernels ----------------

__global__ void k_init(float* amax) { amax[0] = 0.f; amax[1] = 0.f; }

__global__ void k_amax_x(const float* __restrict__ x, float* __restrict__ amax, int n) {
    __shared__ float red[256];
    float m = 0.f;
    for (int i = blockIdx.x * blockDim.x + threadIdx.x; i < n; i += gridDim.x * blockDim.x)
        m = fmaxf(m, fabsf(x[i]));
    red[threadIdx.x] = m;
    __syncthreads();
    for (int s = 128; s > 0; s >>= 1) {
        if (threadIdx.x < s) red[threadIdx.x] = fmaxf(red[threadIdx.x], red[threadIdx.x + s]);
        __syncthreads();
    }
    if (threadIdx.x == 0)
        atomicMax((int*)amax, __float_as_int(red[0]));   // nonneg floats order as ints
}

// per-column amax of W (R x C): thread j = column j (coalesced)
__global__ void k_amax_cols(const float* __restrict__ W, float* __restrict__ amaxv,
                            int R, int C) {
    int j = blockIdx.x * blockDim.x + threadIdx.x;
    if (j >= C) return;
    float m = 1e-8f;
    for (int k = 0; k < R; ++k) m = fmaxf(m, fabsf(W[(size_t)k * C + j]));
    amaxv[j] = m;
}

// per-row amax of W (R x C): one wave32 per row
__global__ void k_amax_rows(const float* __restrict__ W, float* __restrict__ amaxv,
                            int R, int C) {
    int lane = threadIdx.x & 31, wave = threadIdx.x >> 5;
    int r = blockIdx.x * 8 + wave;
    if (r >= R) return;
    float m = 1e-8f;
    for (int k = lane; k < C; k += 32) m = fmaxf(m, fabsf(W[(size_t)r * C + k]));
    for (int off = 16; off; off >>= 1) m = fmaxf(m, __shfl_xor(m, off, 32));
    if (lane == 0) amaxv[r] = m;
}

// tiled transpose + int8 fake-quant: Wt[c*R + r] = q(W[r*C + c]); coalesced both sides
__global__ void k_transq(const float* __restrict__ W, f16* __restrict__ Wt,
                         const float* __restrict__ amaxv, int R, int C, int scale_by_row) {
    __shared__ float tile[32][33];
    int r0 = blockIdx.y * 32, c0 = blockIdx.x * 32;
    int tx = threadIdx.x & 31, ty = threadIdx.x >> 5;    // 32 x 8
    #pragma unroll
    for (int i = 0; i < 32; i += 8)
        tile[ty + i][tx] = W[(size_t)(r0 + ty + i) * C + (c0 + tx)];
    __syncthreads();
    #pragma unroll
    for (int i = 0; i < 32; i += 8) {
        int c = c0 + ty + i;                             // output row  (= original col)
        int r = r0 + tx;                                 // output col  (= original row)
        float m = amaxv[scale_by_row ? r : c];
        float s = 127.f / m, inv = m / 127.f;
        Wt[(size_t)c * R + r] = (f16)fake_q(tile[tx][ty + i], s, inv, -128.f, 127.f);
    }
}

// ---------------- WMMA GEMM 1: hidden = gelu(x_sim @ Win_sim + b_in) ----------------
// Bt = quantized W_in^T, layout [DF][DM]

__global__ __launch_bounds__(256)
void k_gemm1(const float* __restrict__ x, const f16* __restrict__ Bt,
             const float* __restrict__ b_in, f16* __restrict__ hidden,
             float* __restrict__ amax) {
    __shared__ f16 sA[2][BM][LDA];
    __shared__ f16 sB[2][BN][LDA];
    __shared__ float red[256];

    const int t = threadIdx.x, lane = t & 31, wave = t >> 5;
    const int wm = wave >> 1, wn = wave & 1;
    const int m0 = blockIdx.y * BM, n0 = blockIdx.x * BN;
    const int rlow = lane & 15, rsel = lane >> 4;

    const float ax  = fmaxf(amax[0], 1e-8f);
    const float sx  = 32767.f / ax, isx = ax / 32767.f;

    const unsigned sB_base[2] = { lds_off(&sB[0][0][0]), lds_off(&sB[1][0][0]) };

    v8f acc[2][4];
    #pragma unroll
    for (int i = 0; i < 2; ++i)
        #pragma unroll
        for (int j = 0; j < 4; ++j) { v8f z = {}; acc[i][j] = z; }

    float4 Areg[8];

    auto load_A = [&](int kt) {                 // 128x64 f32, 8 float4 per thread
        #pragma unroll
        for (int i = 0; i < 8; ++i) {
            int idx = i * 256 + t;
            int row = idx >> 4, q = idx & 15;
            Areg[i] = *(const float4*)(x + (size_t)(m0 + row) * DM + kt * BK + q * 4);
        }
    };
    auto store_A = [&](int buf) {               // quantize + b64 LDS stores
        #pragma unroll
        for (int i = 0; i < 8; ++i) {
            int idx = i * 256 + t;
            int row = idx >> 4, q = idx & 15;
            v4h h;
            h[0] = (f16)fake_q(Areg[i].x, sx, isx, -32768.f, 32767.f);
            h[1] = (f16)fake_q(Areg[i].y, sx, isx, -32768.f, 32767.f);
            h[2] = (f16)fake_q(Areg[i].z, sx, isx, -32768.f, 32767.f);
            h[3] = (f16)fake_q(Areg[i].w, sx, isx, -32768.f, 32767.f);
            *(v4h*)&sA[buf][row][q * 4] = h;
        }
    };
    auto async_B = [&](int kt, int buf) {       // 128x64 f16 rows: pure async copy
        #pragma unroll
        for (int i = 0; i < 4; ++i) {
            int idx = i * 256 + t;
            int nn = idx >> 3, k8 = idx & 7;
            unsigned lo = sB_base[buf] + (unsigned)(nn * LDA + k8 * 8) * 2u;
            async_b128(lo, Bt + (size_t)(n0 + nn) * DM + kt * BK + k8 * 8);
        }
    };
    auto compute = [&](int buf) {
        #pragma unroll
        for (int s = 0; s < 2; ++s) {
            v16h a[2], b[4];
            #pragma unroll
            for (int mt = 0; mt < 2; ++mt) {
                const f16* p = &sA[buf][wm * 32 + mt * 16 + rlow][s * 32 + rsel * 8];
                v8h lo = *(const v8h*)p;
                v8h hi = *(const v8h*)(p + 16);
                a[mt] = __builtin_shufflevector(lo, hi, 0,1,2,3,4,5,6,7,8,9,10,11,12,13,14,15);
            }
            #pragma unroll
            for (int nt = 0; nt < 4; ++nt) {
                const f16* p = &sB[buf][wn * 64 + nt * 16 + rlow][s * 32 + rsel * 8];
                v8h lo = *(const v8h*)p;
                v8h hi = *(const v8h*)(p + 16);
                b[nt] = __builtin_shufflevector(lo, hi, 0,1,2,3,4,5,6,7,8,9,10,11,12,13,14,15);
            }
            #pragma unroll
            for (int mt = 0; mt < 2; ++mt)
                #pragma unroll
                for (int nt = 0; nt < 4; ++nt)
                    acc[mt][nt] = __builtin_amdgcn_wmma_f32_16x16x32_f16(
                        false, a[mt], false, b[nt], (short)0, acc[mt][nt], false, false);
        }
    };

    const int NK = DM / BK;
    load_A(0); async_B(0, 0); store_A(0);
    wait_async();
    __syncthreads();
    for (int it = 0; it < NK; ++it) {
        int cur = it & 1, nxt = cur ^ 1;
        bool more = (it + 1 < NK);
        if (more) { load_A(it + 1); async_B(it + 1, nxt); }
        compute(cur);
        if (more) {
            store_A(nxt);
            wait_async();
            __syncthreads();
        }
    }

    // epilogue: bias + exact-erf GELU, f16 store, hidden amax
    float lmax = 0.f;
    #pragma unroll
    for (int mt = 0; mt < 2; ++mt) {
        #pragma unroll
        for (int nt = 0; nt < 4; ++nt) {
            int col = n0 + wn * 64 + nt * 16 + rlow;
            float bias = b_in[col];
            #pragma unroll
            for (int r = 0; r < 8; ++r) {
                int row = m0 + wm * 32 + mt * 16 + r + 8 * rsel;
                float v = acc[mt][nt][r] + bias;
                float g = 0.5f * v * (1.f + erff(v * 0.70710678118654752f));
                lmax = fmaxf(lmax, fabsf(g));
                hidden[(size_t)row * DF + col] = (f16)g;
            }
        }
    }
    __syncthreads();
    red[t] = lmax;
    __syncthreads();
    for (int s = 128; s > 0; s >>= 1) {
        if (t < s) red[t] = fmaxf(red[t], red[t + s]);
        __syncthreads();
    }
    if (t == 0) atomicMax((int*)(amax + 1), __float_as_int(red[0]));
}

// ---------------- WMMA GEMM 2: out = hidden_sim @ Wout_sim + b_out ----------------
// Bt = quantized W_out^T, layout [DM][DF]

__global__ __launch_bounds__(256)
void k_gemm2(const f16* __restrict__ hidden, const f16* __restrict__ Bt,
             const float* __restrict__ b_out, float* __restrict__ out,
             const float* __restrict__ amax) {
    __shared__ f16 sA[2][BM][LDA];
    __shared__ f16 sB[2][BN][LDA];

    const int t = threadIdx.x, lane = t & 31, wave = t >> 5;
    const int wm = wave >> 1, wn = wave & 1;
    const int m0 = blockIdx.y * BM, n0 = blockIdx.x * BN;
    const int rlow = lane & 15, rsel = lane >> 4;

    const float ah  = fmaxf(amax[1], 1e-8f);
    const float sh  = 32767.f / ah, ish = ah / 32767.f;

    const unsigned sB_base[2] = { lds_off(&sB[0][0][0]), lds_off(&sB[1][0][0]) };

    v8f acc[2][4];
    #pragma unroll
    for (int i = 0; i < 2; ++i)
        #pragma unroll
        for (int j = 0; j < 4; ++j) { v8f z = {}; acc[i][j] = z; }

    v8h Hreg[4];

    auto load_A = [&](int kt) {                 // 128x64 f16, 4 v8h per thread
        #pragma unroll
        for (int i = 0; i < 4; ++i) {
            int idx = i * 256 + t;
            int row = idx >> 3, q = idx & 7;
            Hreg[i] = *(const v8h*)(hidden + (size_t)(m0 + row) * DF + kt * BK + q * 8);
        }
    };
    auto store_A = [&](int buf) {               // int16 fake-quant + b128 LDS stores
        #pragma unroll
        for (int i = 0; i < 4; ++i) {
            int idx = i * 256 + t;
            int row = idx >> 3, q = idx & 7;
            v8h h;
            #pragma unroll
            for (int e = 0; e < 8; ++e)
                h[e] = (f16)fake_q((float)Hreg[i][e], sh, ish, -32768.f, 32767.f);
            *(v8h*)&sA[buf][row][q * 8] = h;
        }
    };
    auto async_B = [&](int kt, int buf) {
        #pragma unroll
        for (int i = 0; i < 4; ++i) {
            int idx = i * 256 + t;
            int nn = idx >> 3, k8 = idx & 7;
            unsigned lo = sB_base[buf] + (unsigned)(nn * LDA + k8 * 8) * 2u;
            async_b128(lo, Bt + (size_t)(n0 + nn) * DF + kt * BK + k8 * 8);
        }
    };
    auto compute = [&](int buf) {
        #pragma unroll
        for (int s = 0; s < 2; ++s) {
            v16h a[2], b[4];
            #pragma unroll
            for (int mt = 0; mt < 2; ++mt) {
                const f16* p = &sA[buf][wm * 32 + mt * 16 + rlow][s * 32 + rsel * 8];
                v8h lo = *(const v8h*)p;
                v8h hi = *(const v8h*)(p + 16);
                a[mt] = __builtin_shufflevector(lo, hi, 0,1,2,3,4,5,6,7,8,9,10,11,12,13,14,15);
            }
            #pragma unroll
            for (int nt = 0; nt < 4; ++nt) {
                const f16* p = &sB[buf][wn * 64 + nt * 16 + rlow][s * 32 + rsel * 8];
                v8h lo = *(const v8h*)p;
                v8h hi = *(const v8h*)(p + 16);
                b[nt] = __builtin_shufflevector(lo, hi, 0,1,2,3,4,5,6,7,8,9,10,11,12,13,14,15);
            }
            #pragma unroll
            for (int mt = 0; mt < 2; ++mt)
                #pragma unroll
                for (int nt = 0; nt < 4; ++nt)
                    acc[mt][nt] = __builtin_amdgcn_wmma_f32_16x16x32_f16(
                        false, a[mt], false, b[nt], (short)0, acc[mt][nt], false, false);
        }
    };

    const int NK = DF / BK;
    load_A(0); async_B(0, 0); store_A(0);
    wait_async();
    __syncthreads();
    for (int it = 0; it < NK; ++it) {
        int cur = it & 1, nxt = cur ^ 1;
        bool more = (it + 1 < NK);
        if (more) { load_A(it + 1); async_B(it + 1, nxt); }
        compute(cur);
        if (more) {
            store_A(nxt);
            wait_async();
            __syncthreads();
        }
    }

    #pragma unroll
    for (int mt = 0; mt < 2; ++mt) {
        #pragma unroll
        for (int nt = 0; nt < 4; ++nt) {
            int col = n0 + wn * 64 + nt * 16 + rlow;
            float bias = b_out[col];
            #pragma unroll
            for (int r = 0; r < 8; ++r) {
                int row = m0 + wm * 32 + mt * 16 + r + 8 * rsel;
                out[(size_t)row * DM + col] = acc[mt][nt][r] + bias;
            }
        }
    }
}

// ---------------- launch ----------------

extern "C" void kernel_launch(void* const* d_in, const int* in_sizes, int n_in,
                              void* d_out, int out_size, void* d_ws, size_t ws_size,
                              hipStream_t stream) {
    (void)in_sizes; (void)n_in; (void)out_size; (void)ws_size;
    const float* x     = (const float*)d_in[0];
    const float* W_in  = (const float*)d_in[1];
    const float* b_in  = (const float*)d_in[2];
    const float* W_out = (const float*)d_in[3];
    const float* b_out = (const float*)d_in[4];
    float* out = (float*)d_out;

    char* ws = (char*)d_ws;
    float* amax2    = (float*)ws;                               // [0]=amax_x, [1]=amax_hidden
    float* amax_in  = (float*)(ws + 256);                       // per-col of W_in  (4096)
    float* amax_out = (float*)(ws + 256 + 16384);               // per-row of W_out (4096)
    f16* Wq_in_t    = (f16*)(ws + 33280);                       // [DF][DM] f16 (8 MB)
    f16* Wq_out_t   = (f16*)(ws + 33280 + (size_t)DM * DF * 2); // [DM][DF] f16 (8 MB)
    f16* hidden     = (f16*)(ws + 33280 + (size_t)DM * DF * 4); // [NTOK][DF] f16 (64 MB)

    k_init<<<1, 1, 0, stream>>>(amax2);
    k_amax_x<<<512, 256, 0, stream>>>(x, amax2, NTOK * DM);
    k_amax_cols<<<DF / 256, 256, 0, stream>>>(W_in, amax_in, DM, DF);
    k_amax_rows<<<DF / 8, 256, 0, stream>>>(W_out, amax_out, DF, DM);
    k_transq<<<dim3(DF / 32, DM / 32), 256, 0, stream>>>(W_in,  Wq_in_t,  amax_in,  DM, DF, 0);
    k_transq<<<dim3(DM / 32, DF / 32), 256, 0, stream>>>(W_out, Wq_out_t, amax_out, DF, DM, 1);
    k_gemm1<<<dim3(DF / BN, NTOK / BM), 256, 0, stream>>>(x, Wq_in_t, b_in, hidden, amax2);
    k_gemm2<<<dim3(DM / BN, NTOK / BM), 256, 0, stream>>>(hidden, Wq_out_t, b_out, out, amax2);
}
